// AttentionModule_13434657702220
// MI455X (gfx1250) — matrix operand order repaired and checked by reading, not compiled, not run on hardware
//
#include <hip/hip_runtime.h>
#include <hip/hip_bf16.h>
#include <math.h>

// ---------------------------------------------------------------------------
// Problem constants (B=4, H=W=64, C=512)
// ---------------------------------------------------------------------------
#define BB  4
#define HH  64
#define WW  64
#define CC  512
#define HWP 4096          // H*W
#define C2  1024          // 2*C
#define NPIX (BB*HWP)     // 16384
#define MQ  32            // queries per block (2 stacked 16-row subtiles)
#define EPSV 1e-7f

typedef __attribute__((ext_vector_type(16))) __bf16 v16bf;
typedef __attribute__((ext_vector_type(8)))  __bf16 v8bf;
typedef __attribute__((ext_vector_type(8)))  float  v8f;
typedef unsigned short ushort_t;

// float -> bf16 (round to nearest even), stored as raw ushort
__device__ __forceinline__ ushort_t f2bf(float x) {
    unsigned u = __float_as_uint(x);
    unsigned r = (u + 0x7FFFu + ((u >> 16) & 1u)) >> 16;
    return (ushort_t)r;
}

__device__ __forceinline__ v8f wmma_bf16(v16bf a, v16bf b, v8f c) {
    return __builtin_amdgcn_wmma_f32_16x16x32_bf16(false, a, false, b,
                                                   (short)0, c, false, false);
}

__device__ __forceinline__ v16bf pack16(v8bf lo, v8bf hi) {
    return __builtin_shufflevector(lo, hi,
        0,1,2,3,4,5,6,7,8,9,10,11,12,13,14,15);
}

// CDNA5 async global->LDS copy (ASYNCcnt path). VDST = per-lane LDS byte
// address, VADDR = per-lane 64-bit global address.
__device__ __forceinline__ void async_copy_b128(unsigned lds_addr,
                                                const void* gaddr) {
    asm volatile("global_load_async_to_lds_b128 %0, %1, off"
                 :: "v"(lds_addr), "v"(gaddr) : "memory");
}
__device__ __forceinline__ void wait_async0() {
    asm volatile("s_wait_asynccnt 0x0" ::: "memory");
}

// ---------------------------------------------------------------------------
// Kernel 1a: transpose w_comb [2C,C] f32 -> wT [C,2C] bf16
// ---------------------------------------------------------------------------
__global__ __launch_bounds__(256) void prep_w(const float* __restrict__ w,
                                              ushort_t* __restrict__ wT) {
    int idx = blockIdx.x * 256 + threadIdx.x;
    int n = idx >> 10;
    int k = idx & (C2 - 1);
    wT[(size_t)n * C2 + k] = f2bf(w[(size_t)k * CC + n]);
}

// ---------------------------------------------------------------------------
// Kernel 1b: per-pixel prep (one wave per pixel).
//   k  = normalize(f+eps) -> kbf row-major + kT transposed (bf16)
//   g  = (9/cnt) * 3x3 sum-pool of f (pool commutes with the dot product)
//   f  staged into cat[:, C:2C]
// ---------------------------------------------------------------------------
__global__ __launch_bounds__(256) void prep_pixels(const float* __restrict__ f,
                                                   ushort_t* __restrict__ kbf,
                                                   ushort_t* __restrict__ kT,
                                                   ushort_t* __restrict__ gbf,
                                                   ushort_t* __restrict__ cat) {
    const int wave = threadIdx.x >> 5;
    const int lane = threadIdx.x & 31;
    const int pix  = blockIdx.x * 8 + wave;
    const int b = pix >> 12;
    const int p = pix & (HWP - 1);
    const int y = p >> 6, x = p & 63;

    const float* frow = f + (size_t)pix * CC;

    float fe[16];
    float ss = 0.f;
#pragma unroll
    for (int j = 0; j < 16; ++j) {
        int c = lane + 32 * j;
        fe[j] = frow[c] + EPSV;
        ss += fe[j] * fe[j];
    }
#pragma unroll
    for (int off = 16; off >= 1; off >>= 1)
        ss += __shfl_xor(ss, off, 32);
    const float inv = rsqrtf(ss);

    ushort_t* krow = kbf + (size_t)pix * CC;
    ushort_t* kTb  = kT  + (size_t)b * CC * HWP;
    ushort_t* crow = cat + (size_t)pix * C2 + CC;
#pragma unroll
    for (int j = 0; j < 16; ++j) {
        int c = lane + 32 * j;
        ushort_t kb = f2bf(fe[j] * inv);
        krow[c] = kb;
        kTb[(size_t)c * HWP + p] = kb;
        crow[c] = f2bf(frow[c]);
    }

    float acc[16];
#pragma unroll
    for (int j = 0; j < 16; ++j) acc[j] = 0.f;
    int cnt = 0;
    for (int dy = -1; dy <= 1; ++dy)
        for (int dx = -1; dx <= 1; ++dx) {
            int yy = y + dy, xx = x + dx;
            if (yy < 0 || yy >= HH || xx < 0 || xx >= WW) continue;
            ++cnt;
            const float* nr = f + ((size_t)b * HWP + yy * WW + xx) * CC;
#pragma unroll
            for (int j = 0; j < 16; ++j) acc[j] += nr[lane + 32 * j];
        }
    const float sc = 9.0f / (float)cnt;
    ushort_t* grow = gbf + (size_t)pix * CC;
#pragma unroll
    for (int j = 0; j < 16; ++j) grow[lane + 32 * j] = f2bf(acc[j] * sc);
}

// ---------------------------------------------------------------------------
// Kernel 2: fused one-pass flash attention, 32-query tile per block.
// S pass: each wave computes two 16x16 S tiles sharing the same k B-fragment,
// with a distance-2 software pipeline on the B loads.  Online softmax through
// LDS.  Recon pass: kT B-fragments (shared by both row subtiles) and LDS P
// A-fragments are double-buffered one group ahead so L2 latency is covered
// by the previous group's 8 WMMAs.  g tile staged via async global->LDS.
// ---------------------------------------------------------------------------
__global__ __launch_bounds__(256) void attn(const ushort_t* __restrict__ gbf,
                                            const ushort_t* __restrict__ kbf,
                                            const ushort_t* __restrict__ kT,
                                            ushort_t* __restrict__ cat) {
    __shared__ __align__(16) ushort_t lds_g[MQ][CC];          // 32 KB
    __shared__ float lds_max[8][MQ];                          // 1 KB
    __shared__ float lds_sum[8][MQ];                          // 1 KB
    __shared__ __align__(16) ushort_t lds_p[8][2][16][16];    // 8 KB

    const int w    = threadIdx.x >> 5;
    const int lane = threadIdx.x & 31;
    const int half = lane >> 4;
    const int lr   = lane & 15;

    const int b     = blockIdx.x >> 7;               // HW/MQ = 128 tiles/batch
    const int qbase = (blockIdx.x & 127) * MQ;

    const ushort_t* gB   = gbf + (size_t)b * HWP * CC;
    const ushort_t* kB   = kbf + (size_t)b * HWP * CC;
    const ushort_t* kTB  = kT  + (size_t)b * CC * HWP;
    ushort_t*       catB = cat + (size_t)b * HWP * C2;

    // ---- async-stage the 32x512 g tile into LDS (ASYNCcnt path) ----
    {
        const int r   = threadIdx.x >> 3;            // 0..31
        const int seg = (threadIdx.x & 7) * 64;      // column base
        const ushort_t* src = gB + (size_t)(qbase + r) * CC + seg;
        unsigned dst = (unsigned)(size_t)&lds_g[r][seg];
#pragma unroll
        for (int i = 0; i < 8; ++i)
            async_copy_b128(dst + i * 16, src + i * 8);
    }
    wait_async0();
    __syncthreads();

    float m_cur[2][8], l_acc[2][8];
#pragma unroll
    for (int t = 0; t < 2; ++t)
#pragma unroll
        for (int r = 0; r < 8; ++r) { m_cur[t][r] = -INFINITY; l_acc[t][r] = 0.f; }
    v8f O[2][4];
#pragma unroll
    for (int t = 0; t < 2; ++t)
#pragma unroll
        for (int g4 = 0; g4 < 4; ++g4) O[t][g4] = (v8f){0,0,0,0,0,0,0,0};

    for (int it = 0; it < 32; ++it) {
        const int kt = it * 8 + w;
        const int keybase = kt * 16;
        if (it < 31)
            __builtin_prefetch(kB + (size_t)(keybase + 128 + lr) * CC, 0, 1);

        // ---- S tiles: distance-2 pipelined B, A from LDS ----
        const ushort_t* kRow = kB + (size_t)(keybase + lr) * CC + 16 * half;
        v16bf bcur = *(const v16bf*)(kRow);
        v16bf bnx1 = *(const v16bf*)(kRow + 32);
        v8f S[2];
        S[0] = (v8f){0,0,0,0,0,0,0,0};
        S[1] = (v8f){0,0,0,0,0,0,0,0};
#pragma unroll
        for (int s = 0; s < 16; ++s) {
            const int c0 = s * 32;
            v8bf a0lo = *(const v8bf*)&lds_g[lr     ][c0 + 16 * half];
            v8bf a0hi = *(const v8bf*)&lds_g[lr     ][c0 + 16 * half + 8];
            v8bf a1lo = *(const v8bf*)&lds_g[16 + lr][c0 + 16 * half];
            v8bf a1hi = *(const v8bf*)&lds_g[16 + lr][c0 + 16 * half + 8];
            v16bf bnx2 = (s < 14) ? *(const v16bf*)(kRow + c0 + 64) : bcur;
            S[0] = wmma_bf16(pack16(a0lo, a0hi), bcur, S[0]);
            S[1] = wmma_bf16(pack16(a1lo, a1hi), bcur, S[1]);
            bcur = bnx1;
            bnx1 = bnx2;
        }

        // ---- per-row max of both subtiles -> LDS ----
#pragma unroll
        for (int t = 0; t < 2; ++t) {
            float mt[8];
#pragma unroll
            for (int r = 0; r < 8; ++r) {
                float m = S[t][r];
                m = fmaxf(m, __shfl_xor(m, 8, 32));
                m = fmaxf(m, __shfl_xor(m, 4, 32));
                m = fmaxf(m, __shfl_xor(m, 2, 32));
                m = fmaxf(m, __shfl_xor(m, 1, 32));
                mt[r] = m;
            }
            if (lr == 0) {
#pragma unroll
                for (int r = 0; r < 8; ++r)
                    lds_max[w][t * 16 + r + 8 * half] = mt[r];
            }
        }
        __syncthreads();   // also fences last iteration's lds_p reads

        // ---- online-softmax update + P staging ----
#pragma unroll
        for (int t = 0; t < 2; ++t) {
            float mn[8], sc[8];
#pragma unroll
            for (int r = 0; r < 8; ++r) {
                float m = m_cur[t][r];
#pragma unroll
                for (int ww = 0; ww < 8; ++ww)
                    m = fmaxf(m, lds_max[ww][t * 16 + r + 8 * half]);
                mn[r] = m;
                sc[r] = __expf(m_cur[t][r] - m);
                m_cur[t][r] = m;
            }
#pragma unroll
            for (int g4 = 0; g4 < 4; ++g4)
#pragma unroll
                for (int i = 0; i < 8; ++i) O[t][g4][i] *= sc[i];
#pragma unroll
            for (int r = 0; r < 8; ++r) l_acc[t][r] *= sc[r];

            float ps[8];
#pragma unroll
            for (int r = 0; r < 8; ++r) {
                float pv = __expf(S[t][r] - mn[r]);
                ps[r] = pv;
                lds_p[w][t][r + 8 * half][lr] = f2bf(pv);
            }
#pragma unroll
            for (int r = 0; r < 8; ++r) {
                float s = ps[r];
                s += __shfl_xor(s, 8, 32);
                s += __shfl_xor(s, 4, 32);
                s += __shfl_xor(s, 2, 32);
                s += __shfl_xor(s, 1, 32);
                ps[r] = s;
            }
            if (lr == 0) {
#pragma unroll
                for (int r = 0; r < 8; ++r)
                    lds_sum[w][t * 16 + r + 8 * half] = ps[r];
            }
        }
        __syncthreads();
#pragma unroll
        for (int t = 0; t < 2; ++t)
#pragma unroll
            for (int r = 0; r < 8; ++r) {
                float s = 0.f;
#pragma unroll
                for (int ww = 0; ww < 8; ++ww)
                    s += lds_sum[ww][t * 16 + r + 8 * half];
                l_acc[t][r] += s;
            }

        // ---- recon: O += P(16x32).K(32x64), double-buffered one group ----
        v16bf a0c, a1c, bbc[4];
        {   // prime group j=0
            a0c = pack16(*(const v8bf*)&lds_p[half][0][lr][0],
                         *(const v8bf*)&lds_p[half][0][lr][8]);
            a1c = pack16(*(const v8bf*)&lds_p[half][1][lr][0],
                         *(const v8bf*)&lds_p[half][1][lr][8]);
            const int kcol = it * 128 + 16 * half;
#pragma unroll
            for (int gq = 0; gq < 4; ++gq) {
                const int n0 = w * 64 + gq * 16 + lr;
                bbc[gq] = *(const v16bf*)(kTB + (size_t)n0 * HWP + kcol);
            }
        }
#pragma unroll
        for (int j = 0; j < 4; ++j) {
            v16bf a0n, a1n, bbn[4];
            if (j < 3) {        // issue next group's loads before this group's WMMAs
                a0n = pack16(*(const v8bf*)&lds_p[2*j+2+half][0][lr][0],
                             *(const v8bf*)&lds_p[2*j+2+half][0][lr][8]);
                a1n = pack16(*(const v8bf*)&lds_p[2*j+2+half][1][lr][0],
                             *(const v8bf*)&lds_p[2*j+2+half][1][lr][8]);
                const int kcol = it * 128 + (j + 1) * 32 + 16 * half;
#pragma unroll
                for (int gq = 0; gq < 4; ++gq) {
                    const int n0 = w * 64 + gq * 16 + lr;
                    bbn[gq] = *(const v16bf*)(kTB + (size_t)n0 * HWP + kcol);
                }
            } else {
                a0n = a0c; a1n = a1c;
#pragma unroll
                for (int gq = 0; gq < 4; ++gq) bbn[gq] = bbc[gq];
            }
#pragma unroll
            for (int gq = 0; gq < 4; ++gq) {
                O[0][gq] = wmma_bf16(a0c, bbc[gq], O[0][gq]);
                O[1][gq] = wmma_bf16(a1c, bbc[gq], O[1][gq]);
            }
            a0c = a0n; a1c = a1n;
#pragma unroll
            for (int gq = 0; gq < 4; ++gq) bbc[gq] = bbn[gq];
        }
    }

    // ---- epilogue: normalize and write recon into cat[:,0:C] ----
#pragma unroll
    for (int t = 0; t < 2; ++t) {
        float rinv[8];
#pragma unroll
        for (int r = 0; r < 8; ++r)
            rinv[r] = (l_acc[t][r] > 0.f) ? 1.0f / l_acc[t][r] : 0.f;
#pragma unroll
        for (int gq = 0; gq < 4; ++gq) {
            const int col = w * 64 + gq * 16 + lr;
#pragma unroll
            for (int i = 0; i < 8; ++i) {
                const int row = qbase + t * 16 + i + 8 * half;
                catB[(size_t)row * C2 + col] = f2bf(O[t][gq][i] * rinv[i]);
            }
        }
    }
}

// ---------------------------------------------------------------------------
// Kernel 3: combiner GEMM out = cat(16384x1024).wT^T + bias, 32-row tiles,
// B fragments shared by both row subtiles, fully double-buffered K loop.
// ---------------------------------------------------------------------------
__global__ __launch_bounds__(256) void combine(const ushort_t* __restrict__ cat,
                                               const ushort_t* __restrict__ wT,
                                               const float* __restrict__ bias,
                                               float* __restrict__ out) {
    const int w    = threadIdx.x >> 5;
    const int lane = threadIdx.x & 31;
    const int half = lane >> 4;
    const int lr   = lane & 15;
    const int qbase = blockIdx.x * MQ;

    const ushort_t* arow0 = cat + (size_t)(qbase + lr) * C2 + 16 * half;
    const ushort_t* arow1 = cat + (size_t)(qbase + 16 + lr) * C2 + 16 * half;
    const ushort_t* brow  = wT + (size_t)(w * 64 + lr) * C2 + 16 * half;

    v8f O[2][4];
#pragma unroll
    for (int t = 0; t < 2; ++t)
#pragma unroll
        for (int g4 = 0; g4 < 4; ++g4) O[t][g4] = (v8f){0,0,0,0,0,0,0,0};

    v16bf a0c, a1c, bbc[4];
    a0c = pack16(*(const v8bf*)(arow0), *(const v8bf*)(arow0 + 8));
    a1c = pack16(*(const v8bf*)(arow1), *(const v8bf*)(arow1 + 8));
#pragma unroll
    for (int gq = 0; gq < 4; ++gq)
        bbc[gq] = *(const v16bf*)(brow + (size_t)gq * 16 * C2);

#pragma unroll
    for (int s = 0; s < 32; ++s) {
        v16bf a0n, a1n, bbn[4];
        if (s < 31) {           // next K-step's loads in flight over these WMMAs
            const int k0 = (s + 1) * 32;
            a0n = pack16(*(const v8bf*)(arow0 + k0), *(const v8bf*)(arow0 + k0 + 8));
            a1n = pack16(*(const v8bf*)(arow1 + k0), *(const v8bf*)(arow1 + k0 + 8));
#pragma unroll
            for (int gq = 0; gq < 4; ++gq)
                bbn[gq] = *(const v16bf*)(brow + (size_t)gq * 16 * C2 + k0);
        } else {
            a0n = a0c; a1n = a1c;
#pragma unroll
            for (int gq = 0; gq < 4; ++gq) bbn[gq] = bbc[gq];
        }
#pragma unroll
        for (int gq = 0; gq < 4; ++gq) {
            O[0][gq] = wmma_bf16(a0c, bbc[gq], O[0][gq]);
            O[1][gq] = wmma_bf16(a1c, bbc[gq], O[1][gq]);
        }
        a0c = a0n; a1c = a1n;
#pragma unroll
        for (int gq = 0; gq < 4; ++gq) bbc[gq] = bbn[gq];
    }
#pragma unroll
    for (int t = 0; t < 2; ++t)
#pragma unroll
        for (int gq = 0; gq < 4; ++gq) {
            const int col = w * 64 + gq * 16 + lr;
            const float bv = bias[col];
#pragma unroll
            for (int i = 0; i < 8; ++i) {
                const int row = qbase + t * 16 + i + 8 * half;
                out[(size_t)row * CC + col] = O[t][gq][i] + bv;
            }
        }
}

// ---------------------------------------------------------------------------
// Host launcher.  Workspace (~81 MB):
//   [0,16M) k | [16,32M) kT | [32,48M) g | [48,80M) cat | [80,81M) wT
// ---------------------------------------------------------------------------
extern "C" void kernel_launch(void* const* d_in, const int* in_sizes, int n_in,
                              void* d_out, int out_size, void* d_ws, size_t ws_size,
                              hipStream_t stream) {
    const float* f      = (const float*)d_in[0];
    const float* w_comb = (const float*)d_in[2];
    const float* b_comb = (const float*)d_in[3];
    float* out = (float*)d_out;

    char* ws = (char*)d_ws;
    const size_t MB = 1024ull * 1024ull;
    ushort_t* kbf = (ushort_t*)(ws);
    ushort_t* kT  = (ushort_t*)(ws + 16 * MB);
    ushort_t* gbf = (ushort_t*)(ws + 32 * MB);
    ushort_t* cat = (ushort_t*)(ws + 48 * MB);
    ushort_t* wT  = (ushort_t*)(ws + 80 * MB);

    prep_w<<<(CC * C2) / 256, 256, 0, stream>>>(w_comb, wT);
    prep_pixels<<<NPIX / 8, 256, 0, stream>>>(f, kbf, kT, gbf, cat);
    attn<<<NPIX / MQ, 256, 0, stream>>>(gbf, kbf, kT, cat);
    combine<<<NPIX / MQ, 256, 0, stream>>>(cat, wT, b_comb, out);
}
